// MTADGAT_41240275976738
// MI455X (gfx1250) — compile-verified, matrix operand order checked by reading, not run on hardware
//
#include <hip/hip_runtime.h>
#include <hip/hip_bf16.h>
#include <math.h>

typedef __attribute__((ext_vector_type(16))) _Float16 v16h;
typedef __attribute__((ext_vector_type(8)))  _Float16 v8h;
typedef __attribute__((ext_vector_type(8)))  float    v8f;

#define LRELU_SLOPE 0.2f

__device__ __forceinline__ float sigmoidf_(float x) { return 1.0f / (1.0f + expf(-x)); }
__device__ __forceinline__ int imin_(int a, int b) { return a < b ? a : b; }

// Build a 16x16x32 WMMA fragment (A or B) from 32 contiguous f16 at p.
// Lane layout per CDNA5 ISA: lanes 0-15 take k={0..7,16..23}, lanes 16-31 take k={8..15,24..31}.
__device__ __forceinline__ v16h load_frag(const _Float16* p, int sel) {
  v8h lo = *(const v8h*)(p + sel * 8);
  v8h hi = *(const v8h*)(p + 16 + sel * 8);
  v16h r;
#pragma unroll
  for (int i = 0; i < 8; i++) { r[i] = lo[i]; r[i + 8] = hi[i]; }
  return r;
}

// ---------------------------------------------------------------------------
// conv1d (K=64 in/out, ks=7, pad=3) + ReLU.  x:(B,W,K) -> xc:(B,W,K)
// ---------------------------------------------------------------------------
__global__ void conv1d_relu_kernel(const float* __restrict__ x, const float* __restrict__ w,
                                   const float* __restrict__ b, float* __restrict__ xc) {
  int idx = blockIdx.x * blockDim.x + threadIdx.x;  // B*W*K = 262144
  int o = idx & 63;
  int bw = idx >> 6;
  int wp = bw & 255;
  int bb = bw >> 8;
  float acc = b[o];
  const float* wrow = w + o * 64 * 7;
#pragma unroll
  for (int s = 0; s < 7; s++) {
    int ww = wp + s - 3;
    if (ww < 0 || ww >= 256) continue;
    const float* xp = x + (bb * 256 + ww) * 64;
    for (int i = 0; i < 64; i++) acc += xp[i] * wrow[i * 7 + s];
  }
  xc[idx] = fmaxf(acc, 0.0f);
}

// xc (B,W,K) -> xcT (B,K,W)
__global__ void transpose_wk(const float* __restrict__ xc, float* __restrict__ xcT) {
  int idx = blockIdx.x * blockDim.x + threadIdx.x;  // B*K*W
  int w = idx & 255;
  int bk = idx >> 8;
  int k = bk & 63;
  int b = bk >> 6;
  xcT[idx] = xc[(b * 256 + w) * 64 + k];
}

// ---------------------------------------------------------------------------
// Generic f32 GEMM via v_wmma_f32_16x16x32_f16.
// C[M,N] = act(A[M,K] @ op(B) + bias), opB=0: B is KxN (ldb>=N); opB=1: B is NxK (A@B^T).
// Block tile 64x64, 8 wave32s; each wave: 1 M-tile x 2 N-tiles (A-fragment reuse).
// Staging is branchless: addresses are clamped in-bounds, values select-zeroed,
// so loads batch behind a single wait instead of per-load exec-mask branches.
// ---------------------------------------------------------------------------
__global__ __launch_bounds__(256) void gemm_wmma(
    const float* __restrict__ A, const float* __restrict__ Bm,
    const float* __restrict__ bias, float* __restrict__ C,
    int M, int N, int K, int lda, int ldb, int ldc, int opB, int act) {
  __shared__ _Float16 As[64 * 32];
  __shared__ _Float16 Bt[64 * 32];  // transposed: Bt[n][k], n in [0,64)
  int tid = threadIdx.x;
  int lane = tid & 31;
  int wv = tid >> 5;
  int m0 = blockIdx.y * 64, n0 = blockIdx.x * 64;
  int mt = (wv & 3) * 16;           // wave's M sub-tile
  int nh = (wv >> 2) * 32;          // wave's N half (covers 2 tiles of 16)
  int mr = lane & 15, sel = lane >> 4;
  v8f acc0 = {}, acc1 = {};
  for (int kk = 0; kk < K; kk += 32) {
#pragma unroll
    for (int i = 0; i < 8; i++) {  // stage A 64x32 (clamped, branchless)
      int idx = tid + i * 256;
      int r = idx >> 5, c = idx & 31;
      int row = m0 + r, col = kk + c;
      float v = A[(size_t)imin_(row, M - 1) * lda + imin_(col, K - 1)];
      As[r * 32 + c] = (_Float16)((row < M && col < K) ? v : 0.f);
    }
#pragma unroll
    for (int i = 0; i < 8; i++) {  // stage B 64x32 transposed (clamped, branchless)
      int idx = tid + i * 256;
      int n, k;
      float v;
      bool ok;
      if (opB == 0) {
        k = idx >> 6; n = idx & 63;
        ok = (kk + k < K) && (n0 + n < N);
        v = Bm[(size_t)imin_(kk + k, K - 1) * ldb + imin_(n0 + n, N - 1)];
      } else {
        n = idx >> 5; k = idx & 31;
        ok = (n0 + n < N) && (kk + k < K);
        v = Bm[(size_t)imin_(n0 + n, N - 1) * ldb + imin_(kk + k, K - 1)];
      }
      Bt[n * 32 + k] = (_Float16)(ok ? v : 0.f);
    }
    __syncthreads();
    // speculative prefetch of next A tile rows into GL2 (clamped, unconditional)
    __builtin_prefetch(&A[(size_t)imin_(m0 + (tid & 63), M - 1) * lda +
                          imin_(kk + 32, K - 1)], 0, 1);
    v16h af  = load_frag(&As[(mt + mr) * 32], sel);
    v16h bf0 = load_frag(&Bt[(nh + mr) * 32], sel);
    v16h bf1 = load_frag(&Bt[(nh + 16 + mr) * 32], sel);
    acc0 = __builtin_amdgcn_wmma_f32_16x16x32_f16(false, af, false, bf0, (short)0, acc0, false, false);
    acc1 = __builtin_amdgcn_wmma_f32_16x16x32_f16(false, af, false, bf1, (short)0, acc1, false, false);
    __syncthreads();
  }
  int nA = n0 + nh + (lane & 15);
  int nB = nA + 16;
#pragma unroll
  for (int r = 0; r < 8; r++) {
    int m = m0 + mt + r + sel * 8;
    if (m < M && nA < N) {
      float v = acc0[r];
      if (bias) v += bias[nA];
      if (act == 1) v = fmaxf(v, 0.f);
      C[(size_t)m * ldc + nA] = v;
    }
    if (m < M && nB < N) {
      float v = acc1[r];
      if (bias) v += bias[nB];
      if (act == 1) v = fmaxf(v, 0.f);
      C[(size_t)m * ldc + nB] = v;
    }
  }
}

// ---------------------------------------------------------------------------
// GATv2 attention scores: e[b,i,j] = sum_e a[e]*leakyrelu(g1[b,i,e]+g2[b,j,e]) + bias[i,j]
// One wave32 per (b,i,j); wave shfl reduction.
// ---------------------------------------------------------------------------
__global__ void gat_scores(const float* __restrict__ g1, const float* __restrict__ g2,
                           const float* __restrict__ avec, const float* __restrict__ biasmat,
                           float* __restrict__ e, int Nn, int E) {
  int widx = blockIdx.x * 8 + (threadIdx.x >> 5);
  int lane = threadIdx.x & 31;
  int j = widx % Nn;
  int bi = widx / Nn;
  int i = bi % Nn;
  int b = bi / Nn;
  const float* p1 = g1 + (size_t)(b * Nn + i) * E;
  const float* p2 = g2 + (size_t)(b * Nn + j) * E;
  float acc = 0.f;
  for (int ee = lane; ee < E; ee += 32) {
    float s = p1[ee] + p2[ee];
    s = s > 0.f ? s : LRELU_SLOPE * s;
    acc += avec[ee] * s;
  }
#pragma unroll
  for (int off = 16; off > 0; off >>= 1) acc += __shfl_down(acc, off, 32);
  if (lane == 0) e[(size_t)(b * Nn + i) * Nn + j] = acc + biasmat[i * Nn + j];
}

// softmax over 64 + aggregate + sigmoid; writes h_feat transposed back to (B,W,K)
__global__ void gat_agg_feat(const float* __restrict__ e, const float* __restrict__ xn,
                             float* __restrict__ hf) {
  __shared__ float att[64];
  __shared__ float s_max, s_sum;
  int bx = blockIdx.x;           // b*64 + i
  int b = bx >> 6, i = bx & 63;
  int t = threadIdx.x;
  if (t < 64) att[t] = e[(size_t)bx * 64 + t];
  __syncthreads();
  if (t == 0) { float m = att[0]; for (int j = 1; j < 64; j++) m = fmaxf(m, att[j]); s_max = m; }
  __syncthreads();
  if (t < 64) att[t] = expf(att[t] - s_max);
  __syncthreads();
  if (t == 0) { float s = 0.f; for (int j = 0; j < 64; j++) s += att[j]; s_sum = s; }
  __syncthreads();
  float accv = 0.f;  // f = t in [0,256)
  for (int j = 0; j < 64; j++) accv += att[j] * xn[(size_t)(b * 64 + j) * 256 + t];
  hf[(size_t)b * 16384 + t * 64 + i] = sigmoidf_(accv / s_sum);
}

// softmax over 256 + aggregate + sigmoid; h_temp[b,i,f] stays (B,W,K)
__global__ void gat_agg_temp(const float* __restrict__ e, const float* __restrict__ xc,
                             float* __restrict__ ht) {
  __shared__ float att[256];
  __shared__ float red[256];
  __shared__ float s_sum;
  int bx = blockIdx.x;           // b*256 + i
  int b = bx >> 8;
  int t = threadIdx.x;
  float v = e[(size_t)bx * 256 + t];
  red[t] = v;
  __syncthreads();
  for (int s = 128; s > 0; s >>= 1) { if (t < s) red[t] = fmaxf(red[t], red[t + s]); __syncthreads(); }
  float m = red[0];
  __syncthreads();
  float ev = expf(v - m);
  att[t] = ev; red[t] = ev;
  __syncthreads();
  for (int s = 128; s > 0; s >>= 1) { if (t < s) red[t] += red[t + s]; __syncthreads(); }
  if (t == 0) s_sum = red[0];
  __syncthreads();
  int f = t & 63, part = t >> 6;
  float accv = 0.f;
  for (int j = part * 64; j < part * 64 + 64; j++)
    accv += att[j] * xc[(size_t)(b * 256 + j) * 64 + f];
  red[t] = accv;
  __syncthreads();
  if (t < 64) {
    float s = red[t] + red[t + 64] + red[t + 128] + red[t + 192];
    ht[(size_t)bx * 64 + t] = sigmoidf_(s / s_sum);
  }
}

__global__ void concat_kernel(const float* __restrict__ xc, const float* __restrict__ hf,
                              const float* __restrict__ ht, float* __restrict__ hcat) {
  int idx = blockIdx.x * blockDim.x + threadIdx.x;  // 4096*192
  int c = idx % 192;
  int bw = idx / 192;
  float v = (c < 64) ? xc[bw * 64 + c] : (c < 128) ? hf[bw * 64 + c - 64] : ht[bw * 64 + c - 128];
  hcat[idx] = v;
}

// gWh/rWh (450x150 f32) -> zero-padded (464x160) f16 for WMMA B-fragments
__global__ void cvt_wh_f16(const float* __restrict__ src, _Float16* __restrict__ dst) {
  int idx = blockIdx.x * blockDim.x + threadIdx.x;
  if (idx >= 464 * 160) return;
  int n = idx / 160, k = idx % 160;
  float v = src[imin_(n, 449) * 150 + imin_(k, 149)];
  dst[idx] = (n < 450 && k < 150) ? (_Float16)v : (_Float16)0.f;
}

// ---------------------------------------------------------------------------
// Persistent GRU scan: one workgroup (8 wave32s), state in LDS.
// Per step: gh[16,450] = h16[16,150] @ Wh16^T via 29x5 WMMAs (batch=16 == one M tile),
// then fused sigmoid/tanh gate update. Wh16 is L2-resident f16 (464x160 padded).
// ---------------------------------------------------------------------------
__global__ __launch_bounds__(256) void gru_recurrent(
    const float* __restrict__ xw, int xw_b_stride, int xw_t_stride,
    const _Float16* __restrict__ Wh16, const float* __restrict__ bh,
    int T, float* __restrict__ hs_out, float* __restrict__ h_final) {
  __shared__ float h32[2400];          // 16 x 150
  __shared__ _Float16 h16[16 * 160];   // k-padded f16 mirror (A fragments)
  __shared__ float gh[16 * 464];       // n-padded gate pre-acts
  int tid = threadIdx.x, lane = tid & 31, wv = tid >> 5;
  int mr = lane & 15, sel = lane >> 4;
  for (int i = tid; i < 16 * 160; i += 256) h16[i] = (_Float16)0.f;
  for (int i = tid; i < 2400; i += 256) h32[i] = 0.f;
  __syncthreads();
  for (int t = 0; t < T; t++) {
    for (int ntile = wv; ntile < 29; ntile += 8) {   // uniform per wave
      v8f acc = {};
      const _Float16* arow = &h16[mr * 160];
      const _Float16* brow = Wh16 + (size_t)(ntile * 16 + mr) * 160;
#pragma unroll
      for (int ks = 0; ks < 5; ks++) {
        v16h af = load_frag(arow + ks * 32, sel);
        v16h bf = load_frag(brow + ks * 32, sel);
        acc = __builtin_amdgcn_wmma_f32_16x16x32_f16(false, af, false, bf, (short)0, acc,
                                                     false, false);
      }
      int n = ntile * 16 + mr;
#pragma unroll
      for (int r = 0; r < 8; r++) gh[(r + sel * 8) * 464 + n] = acc[r];
    }
    __syncthreads();
    for (int idx = tid; idx < 2400; idx += 256) {
      int b = idx / 150, j = idx - b * 150;
      const float* xwp = xw + (size_t)b * xw_b_stride + (size_t)t * xw_t_stride;
      float hr = gh[b * 464 + j] + bh[j];
      float hz = gh[b * 464 + 150 + j] + bh[150 + j];
      float hn = gh[b * 464 + 300 + j] + bh[300 + j];
      float r = sigmoidf_(xwp[j] + hr);
      float z = sigmoidf_(xwp[150 + j] + hz);
      float nn = tanhf(xwp[300 + j] + r * hn);
      float hnew = (1.f - z) * nn + z * h32[idx];
      h32[idx] = hnew;
      h16[b * 160 + j] = (_Float16)hnew;
      if (hs_out) hs_out[((size_t)b * T + t) * 150 + j] = hnew;
    }
    __syncthreads();
  }
  if (h_final)
    for (int idx = tid; idx < 2400; idx += 256) h_final[idx] = h32[idx];
}

// ---------------------------------------------------------------------------
extern "C" void kernel_launch(void* const* d_in, const int* in_sizes, int n_in,
                              void* d_out, int out_size, void* d_ws, size_t ws_size,
                              hipStream_t stream) {
  (void)in_sizes; (void)n_in; (void)out_size; (void)ws_size;
  const float* x      = (const float*)d_in[0];
  const float* conv_w = (const float*)d_in[1];
  const float* conv_b = (const float*)d_in[2];
  const float* fW     = (const float*)d_in[3];
  const float* fb     = (const float*)d_in[4];
  const float* fa     = (const float*)d_in[5];
  const float* fbias  = (const float*)d_in[6];
  const float* tW     = (const float*)d_in[7];
  const float* tb     = (const float*)d_in[8];
  const float* ta     = (const float*)d_in[9];
  const float* tbias  = (const float*)d_in[10];
  const float* gWi    = (const float*)d_in[11];
  const float* gWh    = (const float*)d_in[12];
  const float* gbi    = (const float*)d_in[13];
  const float* gbh    = (const float*)d_in[14];
  const float* f0w    = (const float*)d_in[15];
  const float* f0b    = (const float*)d_in[16];
  const float* f1w    = (const float*)d_in[17];
  const float* f1b    = (const float*)d_in[18];
  const float* rWi    = (const float*)d_in[19];
  const float* rWh    = (const float*)d_in[20];
  const float* rbi    = (const float*)d_in[21];
  const float* rbh    = (const float*)d_in[22];
  const float* rfw    = (const float*)d_in[23];
  const float* rfb    = (const float*)d_in[24];
  float* out_pred = (float*)d_out;         // 16*64
  float* out_rec  = (float*)d_out + 1024;  // 16*256*64

  float* ws = (float*)d_ws;
  size_t off = 0;
  auto alloc = [&](size_t n) { float* p = ws + off; off += (n + 3) & ~(size_t)3; return p; };
  float* xc      = alloc(262144);
  float* xcT     = alloc(262144);
  float* g1f     = alloc(524288);
  float* g2f     = alloc(524288);
  float* e_f     = alloc(65536);
  float* h_feat  = alloc(262144);
  float* g1t     = alloc(524288);
  float* g2t     = alloc(524288);
  float* e_t     = alloc(1048576);
  float* h_temp  = alloc(262144);
  float* hcat    = alloc(786432);
  float* xw_enc  = alloc(1843200);
  float* xw_dec  = alloc(7200);
  float* h_end   = alloc(2400);
  float* tmpfc   = alloc(2400);
  float* dec_out = alloc(614400);
  _Float16* gWh16 = (_Float16*)alloc(37120);
  _Float16* rWh16 = (_Float16*)alloc(37120);

  conv1d_relu_kernel<<<1024, 256, 0, stream>>>(x, conv_w, conv_b, xc);
  transpose_wk<<<1024, 256, 0, stream>>>(xc, xcT);

  // feature-axis GATv2: nodes=64, feat=256, E=512
  dim3 fgrid(512 / 64, 1024 / 64);
  gemm_wmma<<<fgrid, 256, 0, stream>>>(xcT, fW,             nullptr, g1f, 1024, 512, 256, 256, 512, 512, 0, 0);
  gemm_wmma<<<fgrid, 256, 0, stream>>>(xcT, fW + 256 * 512, fb,      g2f, 1024, 512, 256, 256, 512, 512, 0, 0);
  gat_scores<<<(16 * 64 * 64) / 8, 256, 0, stream>>>(g1f, g2f, fa, fbias, e_f, 64, 512);
  gat_agg_feat<<<1024, 256, 0, stream>>>(e_f, xcT, h_feat);

  // time-axis GATv2: nodes=256, feat=64, E=128
  dim3 tgrid(2, 4096 / 64);
  gemm_wmma<<<tgrid, 256, 0, stream>>>(xc, tW,            nullptr, g1t, 4096, 128, 64, 64, 128, 128, 0, 0);
  gemm_wmma<<<tgrid, 256, 0, stream>>>(xc, tW + 64 * 128, tb,      g2t, 4096, 128, 64, 64, 128, 128, 0, 0);
  gat_scores<<<(16 * 256 * 256) / 8, 256, 0, stream>>>(g1t, g2t, ta, tbias, e_t, 256, 128);
  gat_agg_temp<<<4096, 256, 0, stream>>>(e_t, xc, h_temp);

  // encoder GRU
  concat_kernel<<<3072, 256, 0, stream>>>(xc, h_feat, h_temp, hcat);
  dim3 xwgrid((450 + 63) / 64, 4096 / 64);
  gemm_wmma<<<xwgrid, 256, 0, stream>>>(hcat, gWi, gbi, xw_enc, 4096, 450, 192, 192, 192, 450, 1, 0);
  cvt_wh_f16<<<290, 256, 0, stream>>>(gWh, gWh16);
  cvt_wh_f16<<<290, 256, 0, stream>>>(rWh, rWh16);
  gru_recurrent<<<1, 256, 0, stream>>>(xw_enc, 256 * 450, 450, gWh16, gbh, 256, nullptr, h_end);

  // prediction head
  dim3 p1grid((150 + 63) / 64, 1);
  gemm_wmma<<<p1grid, 256, 0, stream>>>(h_end, f0w, f0b, tmpfc, 16, 150, 150, 150, 150, 150, 0, 1);
  dim3 p2grid(1, 1);
  gemm_wmma<<<p2grid, 256, 0, stream>>>(tmpfc, f1w, f1b, out_pred, 16, 64, 150, 150, 64, 64, 0, 0);

  // decoder GRU (constant input => xw computed once) + reconstruction
  dim3 dgrid((450 + 63) / 64, 1);
  gemm_wmma<<<dgrid, 256, 0, stream>>>(h_end, rWi, rbi, xw_dec, 16, 450, 150, 150, 150, 450, 1, 0);
  gru_recurrent<<<1, 256, 0, stream>>>(xw_dec, 450, 0, rWh16, rbh, 256, dec_out, nullptr);
  dim3 rgrid(1, 4096 / 64);
  gemm_wmma<<<rgrid, 256, 0, stream>>>(dec_out, rfw, rfb, out_rec, 4096, 64, 150, 150, 64, 64, 0, 0);
}
